// Net_CCL_69277822485257
// MI455X (gfx1250) — matrix operand order repaired and checked by reading, not compiled, not run on hardware
//
#include <hip/hip_runtime.h>
#include <hip/hip_bf16.h>
#include <math.h>

// ---------------------------------------------------------------------------
// FFT-domain circular conv net == direct circular conv (<=21 taps) -> implicit
// GEMM on bf16 WMMA (v_wmma_f32_16x16x32_bf16), f32 accumulate.
// Activations: channels-last bf16, Ci padded to 64 (pad channels = 0).
// Weights: pre-swizzled into WMMA A fragments, contiguous k = 16*(lane/16) + e.
// All spatial dims are powers of two -> circular wrap is a single AND.
// ---------------------------------------------------------------------------

typedef __attribute__((ext_vector_type(16))) __bf16 v16bf;
typedef __attribute__((ext_vector_type(8)))  float  v8f;

// ------------------------- weight pre-swizzle (A-matrix) -------------------
// afrag layout: [4 m-tiles][KCH=KH*KW*2 chunks][32 lanes][16 elems] bf16
// A k order used here (contiguous): lanes 0-15: K(e)= e
//                        lanes16-31: K(e)= e+16
__global__ void prep_a_kernel(const float* __restrict__ w, __bf16* __restrict__ afrag,
                              int Co, int Ci, int KH, int KW) {
    int KCH = KH * KW * 2;
    int total = 4 * KCH * 512;
    int idx = blockIdx.x * blockDim.x + threadIdx.x;
    if (idx >= total) return;
    int e    = idx & 15;
    int lane = (idx >> 4) & 31;
    int kc   = (idx >> 9) % KCH;
    int mt   = (idx >> 9) / KCH;
    int k    = (lane >> 4) * 16 + e;
    int ci   = (kc & 1) * 32 + k;
    int tap  = kc >> 1;
    int p = tap / KW, q = tap % KW;
    int m = mt * 16 + (lane & 15);
    float v = 0.0f;
    if (m < Co && ci < Ci) v = w[((m * Ci + ci) * KH + p) * KW + q];
    afrag[idx] = (__bf16)v;
}

// ------------------------- generic zero fill -------------------------------
__global__ void zero_words_kernel(unsigned int* __restrict__ p, long long nwords) {
    long long i = (long long)blockIdx.x * blockDim.x + threadIdx.x;
    long long stride = (long long)gridDim.x * blockDim.x;
    for (; i < nwords; i += stride) p[i] = 0u;
}

// ------------------------- layer 1: Ci=1 direct conv + pool(2,4) + relu ----
// x: [32,64,256] f32 ; out buf1: [32,32,64,64ch] bf16 (ch 50..63 = 0)
__global__ void layer1_kernel(const float* __restrict__ x, const float* __restrict__ w1,
                              const float* __restrict__ b1, __bf16* __restrict__ out) {
    int idx = blockIdx.x * blockDim.x + threadIdx.x;
    if (idx >= 32 * 32 * 64 * 64) return;
    int ch = idx & 63;
    int pw = (idx >> 6) & 63;
    int ph = (idx >> 12) & 31;
    int b  = idx >> 17;
    __bf16 res = (__bf16)0.0f;
    if (ch < 50) {
        const float* wk = w1 + ch * 21;              // [50,1,3,7]
        const float* xb = x + (long long)b * 64 * 256;
        float mx = -3.0e38f;
        for (int di = 0; di < 2; ++di) {
            for (int dj = 0; dj < 4; ++dj) {
                int i = ph * 2 + di, j = pw * 4 + dj;
                float acc = b1[ch];
#pragma unroll
                for (int p = 0; p < 3; ++p) {
                    int ii = (i - p) & 63;
#pragma unroll
                    for (int q = 0; q < 7; ++q) {
                        int jj = (j - q) & 255;
                        acc += xb[ii * 256 + jj] * wk[p * 7 + q];
                    }
                }
                mx = fmaxf(mx, acc);
            }
        }
        res = (__bf16)fmaxf(mx, 0.0f);
    }
    out[idx] = res;
}

// ------------------------- WMMA circular-conv implicit GEMM ----------------
// xin : [nimg, H, W, 64ch] bf16 (pad ch zero).  afrag: pre-swizzled weights.
// H, W compile-time powers of two -> wrap via AND, strip mapping via shifts.
// MODE 0: write raw conv+bias f32 -> [nimg,H,W,64] (pre-pool)
// MODE 1: relu -> bf16 [nimg,H,W,64] (pad ch written as 0)
// MODE 2: relu -> bf16 scattered to zero-upsampled [nimg,2H,4W,64] at (2i,4j)
template <int KH, int KW, int H, int W, int MODE>
__global__ __launch_bounds__(128) void conv_wmma_kernel(
    const __bf16* __restrict__ xin, const __bf16* __restrict__ afrag,
    const float* __restrict__ bias, void* __restrict__ out, int Co) {
    constexpr int KCH = KH * KW * 2;
    constexpr int SPR = W / 64;          // strips per row (power of two)
    const int lane = threadIdx.x & 31;
    const int wave = threadIdx.x >> 5;   // M-tile id (0..3)
    const int img  = blockIdx.y;
    const int strip = blockIdx.x;        // 64 consecutive pixels in one row
    const int i  = strip / SPR;
    const int j0 = (strip % SPR) * 64;
    const int n  = lane & 15;
    const int hi = lane >> 4;            // lane half

    // hoisted circularly-wrapped row base pointers (p = 0..KH-1)
    const __bf16* rowptr[KH];
#pragma unroll
    for (int p = 0; p < KH; ++p) {
        int ii = (i - p) & (H - 1);
        rowptr[p] = xin + ((long long)img * H + ii) * W * 64;
    }

    v8f acc[4] = {};
    const __bf16* aw = afrag + (long long)wave * KCH * 512 + lane * 16;
    // B k order used here (same as A): lanes 0-15 hold K 0..15, lanes 16-31 hold K 16..31
    const int cihalf = hi * 16;

#pragma unroll
    for (int kc = 0; kc < KCH; ++kc) {
        v16bf a = *(const v16bf*)(aw + kc * 512);
        constexpr int unused = 0; (void)unused;
        const int tap = kc >> 1;
        const int p = tap / KW, q = tap % KW;
        const int ci0 = (kc & 1) * 32 + cihalf;
        const __bf16* rb = rowptr[p];
#pragma unroll
        for (int s = 0; s < 4; ++s) {
            int jj = (j0 + s * 16 + n - q) & (W - 1);   // circular wrap
            v16bf bm = *(const v16bf*)(rb + jj * 64 + ci0);
            acc[s] = __builtin_amdgcn_wmma_f32_16x16x32_bf16(
                false, a, false, bm, (short)0, acc[s], false, false);
        }
    }

    // C/D layout: VGPR r -> M = r (lanes 0-15) / r+8 (lanes 16-31), N = lane&15
    // A-fragment rows >= Co are zero-padded, so acc==0 there; zero-guarded bias
    // keeps the epilogue branchless and writes all 64 channels (deterministic).
    const int m0 = wave * 16 + hi * 8;
    float bv[8];
#pragma unroll
    for (int r = 0; r < 8; ++r) bv[r] = (m0 + r < Co) ? bias[m0 + r] : 0.0f;

#pragma unroll
    for (int s = 0; s < 4; ++s) {
        const int j = j0 + s * 16 + n;
#pragma unroll
        for (int r = 0; r < 8; ++r) {
            const int m = m0 + r;
            const float v = acc[s][r] + bv[r];
            if (MODE == 0) {
                ((float*)out)[(((long long)img * H + i) * W + j) * 64 + m] = v;
            } else if (MODE == 1) {
                ((__bf16*)out)[(((long long)img * H + i) * W + j) * 64 + m] =
                    (__bf16)fmaxf(v, 0.0f);
            } else {
                constexpr int H2 = H * 2, W2 = W * 4;
                ((__bf16*)out)[(((long long)img * H2 + 2 * i) * W2 + 4 * j) * 64 + m] =
                    (__bf16)fmaxf(v, 0.0f);
            }
        }
    }
}

// ------------------------- pool(2,4)+relu after layer 2 --------------------
// pre2: [32,32,64,64] f32 -> buf2: [32,16,16,64] bf16
__global__ void pool2_kernel(const float* __restrict__ pre, __bf16* __restrict__ out) {
    int idx = blockIdx.x * blockDim.x + threadIdx.x;
    if (idx >= 32 * 16 * 16 * 64) return;
    int ch = idx & 63;
    int pw = (idx >> 6) & 15;
    int ph = (idx >> 10) & 15;
    int b  = idx >> 14;
    __bf16 res = (__bf16)0.0f;
    if (ch < 50) {
        float mx = -3.0e38f;
#pragma unroll
        for (int di = 0; di < 2; ++di)
#pragma unroll
            for (int dj = 0; dj < 4; ++dj)
                mx = fmaxf(mx, pre[(((long long)b * 32 + ph * 2 + di) * 64 + pw * 4 + dj) * 64 + ch]);
        res = (__bf16)fmaxf(mx, 0.0f);
    }
    out[idx] = res;
}

// ------------------------- pair mean over 16x16 (h[:,1]) -------------------
__global__ void mean_kernel(const __bf16* __restrict__ buf2, float* __restrict__ meanb) {
    int idx = blockIdx.x * blockDim.x + threadIdx.x;   // 16*64
    if (idx >= 16 * 64) return;
    int ch = idx & 63, b = idx >> 6;
    const __bf16* src = buf2 + (((long long)(b * 2 + 1) * 16) * 16) * 64 + ch;
    float s = 0.0f;
    for (int t = 0; t < 256; ++t) s += (float)src[(long long)t * 64];
    meanb[idx] = s * (1.0f / 256.0f);
}

// ------------------------- combine + zero-upsample(2,4) scatter ------------
// ups3 pre-zeroed [16,32,64,64] bf16; write at (2i,4j)
__global__ void combine_kernel(const __bf16* __restrict__ buf2, const float* __restrict__ meanb,
                               __bf16* __restrict__ ups3) {
    int idx = blockIdx.x * blockDim.x + threadIdx.x;   // 16*16*16*64
    if (idx >= 16 * 16 * 16 * 64) return;
    int ch = idx & 63;
    int j  = (idx >> 6) & 15;
    int i  = (idx >> 10) & 15;
    int b  = idx >> 14;
    float v = (float)buf2[((((long long)(b * 2) * 16) + i) * 16 + j) * 64 + ch] + meanb[b * 64 + ch];
    ups3[(((long long)b * 32 + 2 * i) * 64 + 4 * j) * 64 + ch] = (__bf16)v;
}

// ------------------------- layer 5: Co=1 direct conv + sigmoid -------------
// buf4: [16,64,256,64] bf16 -> out [16,64,256] f32
__global__ void layer5_kernel(const __bf16* __restrict__ xin, const float* __restrict__ w5,
                              const float* __restrict__ b5, float* __restrict__ out) {
    int idx = blockIdx.x * blockDim.x + threadIdx.x;
    if (idx >= 16 * 64 * 256) return;
    int j = idx & 255;
    int i = (idx >> 8) & 63;
    int b = idx >> 14;
    float acc = b5[0];
#pragma unroll
    for (int p = 0; p < 3; ++p) {
        int ii = (i - p) & 63;
#pragma unroll
        for (int q = 0; q < 7; ++q) {
            int jj = (j - q) & 255;
            const __bf16* src = xin + (((long long)b * 64 + ii) * 256 + jj) * 64;
            const float* wk = w5 + p * 7 + q;        // w5[(ci*3+p)*7+q]
            for (int ci = 0; ci < 50; ++ci)
                acc += (float)src[ci] * wk[ci * 21];
        }
    }
    out[idx] = 1.0f / (1.0f + __expf(-acc));
}

// ---------------------------------------------------------------------------
extern "C" void kernel_launch(void* const* d_in, const int* in_sizes, int n_in,
                              void* d_out, int out_size, void* d_ws, size_t ws_size,
                              hipStream_t stream) {
    (void)in_sizes; (void)n_in; (void)out_size; (void)ws_size;
    const float* x  = (const float*)d_in[0];
    const float* w1 = (const float*)d_in[1];  const float* b1 = (const float*)d_in[2];
    const float* w2 = (const float*)d_in[3];  const float* b2 = (const float*)d_in[4];
    const float* w3 = (const float*)d_in[5];  const float* b3 = (const float*)d_in[6];
    const float* w4 = (const float*)d_in[7];  const float* b4 = (const float*)d_in[8];
    const float* w5 = (const float*)d_in[9];  const float* b5 = (const float*)d_in[10];
    float* out = (float*)d_out;

    // bump allocator over workspace (total ~98 MB)
    char* base = (char*)d_ws;
    size_t off = 0;
    auto alloc = [&](size_t bytes) { size_t r = off; off = (off + bytes + 255) & ~(size_t)255; return base + r; };
    __bf16* buf1  = (__bf16*)alloc((size_t)32 * 32 * 64  * 64 * 2);  // L1 out pooled
    float*  pre2  = (float*) alloc((size_t)32 * 32 * 64  * 64 * 4);  // L2 pre-pool
    __bf16* buf2  = (__bf16*)alloc((size_t)32 * 16 * 16  * 64 * 2);  // L2 pooled
    float*  meanb = (float*) alloc((size_t)16 * 64 * 4);
    __bf16* ups3  = (__bf16*)alloc((size_t)16 * 32 * 64  * 64 * 2);  // L3 input (upsampled)
    __bf16* ups4  = (__bf16*)alloc((size_t)16 * 64 * 256 * 64 * 2);  // L4 input (upsampled)
    __bf16* buf4  = (__bf16*)alloc((size_t)16 * 64 * 256 * 64 * 2);  // L4 out
    __bf16* a2    = (__bf16*)alloc((size_t)4 * 42 * 512 * 2);
    __bf16* a3    = (__bf16*)alloc((size_t)4 * 30 * 512 * 2);
    __bf16* a4    = (__bf16*)alloc((size_t)4 * 42 * 512 * 2);

    // weight swizzle into WMMA A-fragment layout
    prep_a_kernel<<<(4 * 42 * 512 + 255) / 256, 256, 0, stream>>>(w2, a2, 50, 50, 3, 7);
    prep_a_kernel<<<(4 * 30 * 512 + 255) / 256, 256, 0, stream>>>(w3, a3, 50, 50, 3, 5);
    prep_a_kernel<<<(4 * 42 * 512 + 255) / 256, 256, 0, stream>>>(w4, a4, 50, 50, 3, 7);

    // zero the zero-upsampled buffers (scatter targets) every call
    zero_words_kernel<<<1024, 256, 0, stream>>>((unsigned int*)ups3, (long long)(16 * 32 * 64 * 64 * 2) / 4);
    zero_words_kernel<<<2048, 256, 0, stream>>>((unsigned int*)ups4, (long long)16 * 64 * 256 * 64 * 2 / 4);

    // L1: direct conv (Ci=1) + pool(2,4) + relu -> channels-last bf16
    layer1_kernel<<<(32 * 32 * 64 * 64 + 255) / 256, 256, 0, stream>>>(x, w1, b1, buf1);

    // L2: WMMA conv 3x7 on [32,32,64] -> f32 pre-pool
    conv_wmma_kernel<3, 7, 32, 64, 0><<<dim3(32, 32), 128, 0, stream>>>(buf1, a2, b2, pre2, 50);
    pool2_kernel<<<(32 * 16 * 16 * 64 + 255) / 256, 256, 0, stream>>>(pre2, buf2);

    // pair combine: h[:,0] + mean(h[:,1]) -> zero-upsample into ups3
    mean_kernel<<<(16 * 64 + 255) / 256, 256, 0, stream>>>(buf2, meanb);
    combine_kernel<<<(16 * 16 * 16 * 64 + 255) / 256, 256, 0, stream>>>(buf2, meanb, ups3);

    // L3: WMMA conv 3x5 on [16,32,64], relu, scatter into zero-upsampled ups4
    conv_wmma_kernel<3, 5, 32, 64, 2><<<dim3(32, 16), 128, 0, stream>>>(ups3, a3, b3, ups4, 50);

    // L4: WMMA conv 3x7 on [16,64,256], relu -> buf4
    conv_wmma_kernel<3, 7, 64, 256, 1><<<dim3(256, 16), 128, 0, stream>>>(ups4, a4, b4, buf4, 50);

    // L5: Co=1 direct conv + sigmoid -> output
    layer5_kernel<<<(16 * 64 * 256 + 255) / 256, 256, 0, stream>>>(buf4, w5, b5, out);
}
